// ParticleEnergyPredictorHuang_10144712753707
// MI455X (gfx1250) — matrix operand order repaired and checked by reading, not compiled, not run on hardware
//
#include <hip/hip_runtime.h>

// ---------------------------------------------------------------------------
// ParticleEnergyPredictor fwd+bwd for MI455X (gfx1250, wave32, WMMA).
// Round 3: address-space-qualified fragment loaders (ds_load_b128 /
// global_load_b128 without generic-pointer null checks) and branchless
// hardware transcendentals (__expf-based tanh, __sinf/__cosf/__powf).
// ---------------------------------------------------------------------------

typedef _Float16 half_t;
typedef __attribute__((ext_vector_type(16))) _Float16 v16h;
typedef __attribute__((ext_vector_type(8)))  _Float16 v8h;
typedef __attribute__((ext_vector_type(8)))  float    v8f;
typedef __attribute__((ext_vector_type(4)))  float    v4f;

#define ASL __attribute__((address_space(3)))   // LDS
#define ASG __attribute__((address_space(1)))   // global

typedef ASL half_t* lds_hp;
typedef const ASL half_t* lds_chp;
typedef ASG half_t* glb_hp;
typedef const ASG half_t* glb_chp;
typedef ASG float* glb_fp;
typedef const ASG float* glb_cfp;

static constexpr float PI_F  = 3.14159265358979323846f;
static constexpr float EPSF  = 1e-8f;
static constexpr float RCUT  = 2.5f;
static constexpr float SLOPE = 0.01f;   // jax.nn.leaky_relu default

// ---- workspace layout (bytes) ----
static constexpr size_t OFF_PW0   = 0;         // [128][32]  f16 packed B (pw0)
static constexpr size_t OFF_PW1   = 8192;      // [128][128] f16 packed B (pw1)
static constexpr size_t OFF_PW1T  = 40960;     // [128][128] f16 packed B (pw1^T)
static constexpr size_t OFF_PW0T  = 73728;     // [16][128]  f16 packed B (pw0^T)
static constexpr size_t OFF_EW0   = 77824;     // [256][256] f16 packed B (ew0)
static constexpr size_t OFF_EW1   = 208896;    // [256][256] f16 packed B (ew1)
static constexpr size_t OFF_EW1T  = 339968;    // [256][256] f16 packed B (ew1^T)
static constexpr size_t OFF_EW0T  = 471040;    // [256][256] f16 packed B (ew0^T)
static constexpr size_t OFF_H0E   = 602112;    // 1024x256 f16
static constexpr size_t OFF_H1E   = 1126400;   // 1024x256 f16
static constexpr size_t OFF_GZ0E  = 1650688;   // 1024x256 f16
static constexpr size_t OFF_DES   = 2174976;   // 1024x256 f32 (240 used)
static constexpr size_t OFF_GDES  = 3223552;   // 1024x256 f32 (240 used)
static constexpr size_t OFF_PS    = 4272128;   // 1024 f32 prior_sum
static constexpr size_t OFF_GPS   = 4276224;   // 1024 f32 dE/dprior_sum
static constexpr size_t OFF_GS1   = 4280320;   // 1024 f32 dE/d(mlp_e out)
static constexpr size_t OFF_GORI  = 4284416;   // 1024x9 f32 g_orientation

// ---------------------------------------------------------------------------
// Math helpers (branchless, hardware transcendentals)
// ---------------------------------------------------------------------------
__device__ __forceinline__ float fast_tanh(float x) {
  x = fminf(fmaxf(x, -15.f), 15.f);             // tanh saturated anyway
  float e = __expf(2.f * x);                    // v_exp_f32 path
  return (e - 1.f) * __builtin_amdgcn_rcpf(e + 1.f);
}

// ---------------------------------------------------------------------------
// WMMA helpers
// ---------------------------------------------------------------------------
__device__ __forceinline__ v8f vzero8() {
  v8f c = {0.f, 0.f, 0.f, 0.f, 0.f, 0.f, 0.f, 0.f};
  return c;
}

__device__ __forceinline__ v8f wmma16(v16h a, v16h b, v8f c) {
  return __builtin_amdgcn_wmma_f32_16x16x32_f16(false, a, false, b, (short)0, c,
                                                false, false);
}

__device__ __forceinline__ v16h cat8(v8h lo, v8h up) {
  return __builtin_shufflevector(lo, up, 0, 1, 2, 3, 4, 5, 6, 7, 8, 9, 10, 11,
                                 12, 13, 14, 15);
}

// A fragment from row-major f16 LDS [M][ld]: per-lane halves are K runs
// [k0..k0+7] and [k0+16..k0+23], k0 = (lane>=16)*8. Two ds_load_b128.
__device__ __forceinline__ v16h loadA_lds(lds_chp src, int ld, int mBase,
                                          int kBase, int lane) {
  const int hi = lane >> 4;
  lds_chp p = src + (mBase + (lane & 15)) * ld + kBase + (hi ? 8 : 0);
  v8h lo = *(const ASL v8h*)p;
  v8h up = *(const ASL v8h*)(p + 16);
  return cat8(lo, up);
}

// Same layout, global f16 source. Two global_load_b128.
__device__ __forceinline__ v16h loadA_g16(glb_chp src, int ld, int mBase,
                                          int kBase, int lane) {
  const int hi = lane >> 4;
  glb_chp p = src + (size_t)(mBase + (lane & 15)) * ld + kBase + (hi ? 8 : 0);
  v8h lo = *(const ASG v8h*)p;
  v8h up = *(const ASG v8h*)(p + 16);
  return cat8(lo, up);
}

// Same layout, global f32 source (DES): 4x global_load_b128 + cvt.
__device__ __forceinline__ v16h loadA_gf32(glb_cfp src, int ld, int mBase,
                                           int kBase, int lane) {
  const int hi = lane >> 4;
  glb_cfp p = src + (size_t)(mBase + (lane & 15)) * ld + kBase + (hi ? 8 : 0);
  v4f a0 = *(const ASG v4f*)p;
  v4f a1 = *(const ASG v4f*)(p + 4);
  v4f a2 = *(const ASG v4f*)(p + 16);
  v4f a3 = *(const ASG v4f*)(p + 20);
  v16h a;
#pragma unroll
  for (int i = 0; i < 4; ++i) {
    a[i] = (half_t)a0[i];
    a[4 + i] = (half_t)a1[i];
    a[8 + i] = (half_t)a2[i];
    a[12 + i] = (half_t)a3[i];
  }
  return a;
}

// B fragment from packed [N][K] f16 global buffer: lane's halves are K runs
// [k0..k0+7] and [k0+16..k0+23], k0 = (lane>=16)*8, at column (nBase + lane%16).
__device__ __forceinline__ v16h loadB_pk(glb_chp srcNK, int ldk, int kBase,
                                         int nBase, int lane) {
  glb_chp p = srcNK + (size_t)(nBase + (lane & 15)) * ldk + kBase +
              ((lane >> 4) << 3);
  v8h lo = *(const ASG v8h*)p;
  v8h up = *(const ASG v8h*)(p + 16);
  return cat8(lo, up);
}

// ---------------------------------------------------------------------------
// Shared prior-MLP forward: smF[64x32] -> smH0[64x128] -> smH1[64x128] (tanh).
// 8 waves: wave -> M-tile (wave>>1), N-tiles ((wave&1)*4 .. +3).
// ---------------------------------------------------------------------------
__device__ void prior_mlp_fwd(lds_chp smF, lds_hp smH0, lds_hp smH1,
                              glb_chp wPW0, glb_chp wPW1, const float* pb0,
                              const float* pb1) {
  const int t = threadIdx.x, lane = t & 31, wave = t >> 5;
  const int mt = wave >> 1, ntB = (wave & 1) * 4;
  {  // layer 0: [64x32]@[32x128]
    v8f acc[4] = {vzero8(), vzero8(), vzero8(), vzero8()};
    v16h a = loadA_lds(smF, 32, 16 * mt, 0, lane);
#pragma unroll
    for (int j = 0; j < 4; ++j) {
      v16h bf = loadB_pk(wPW0, 32, 0, 16 * (ntB + j), lane);
      acc[j] = wmma16(a, bf, acc[j]);
    }
#pragma unroll
    for (int j = 0; j < 4; ++j) {
      int nn = 16 * (ntB + j) + (lane & 15);
      float bias = pb0[nn];
#pragma unroll
      for (int v = 0; v < 8; ++v) {
        int m = 16 * mt + v + 8 * (lane >> 4);
        smH0[m * 128 + nn] = (half_t)fast_tanh(acc[j][v] + bias);
      }
    }
  }
  __syncthreads();
  {  // layer 1: [64x128]@[128x128], K in 4 chunks of 32
    v8f acc[4] = {vzero8(), vzero8(), vzero8(), vzero8()};
#pragma unroll
    for (int kc = 0; kc < 4; ++kc) {
      v16h a = loadA_lds((lds_chp)smH0, 128, 16 * mt, 32 * kc, lane);
#pragma unroll
      for (int j = 0; j < 4; ++j) {
        v16h bf = loadB_pk(wPW1, 128, 32 * kc, 16 * (ntB + j), lane);
        acc[j] = wmma16(a, bf, acc[j]);
      }
    }
#pragma unroll
    for (int j = 0; j < 4; ++j) {
      int nn = 16 * (ntB + j) + (lane & 15);
      float bias = pb1[nn];
#pragma unroll
      for (int v = 0; v < 8; ++v) {
        int m = 16 * mt + v + 8 * (lane >> 4);
        smH1[m * 128 + nn] = (half_t)fast_tanh(acc[j][v] + bias);
      }
    }
  }
  __syncthreads();
}

// ---------------------------------------------------------------------------
// Kernel: zero accumulators (d_out force/torque, ws DES/prior_sum/g_ori)
// ---------------------------------------------------------------------------
__global__ __launch_bounds__(256) void k_zero(float* outFT, float* wsDES,
                                              float* wsPS, float* wsGORI) {
  int idx = blockIdx.x * 256 + threadIdx.x;
  if (idx < 6144) outFT[idx] = 0.f;
  if (idx < 262144) wsDES[idx] = 0.f;
  if (idx < 1024) wsPS[idx] = 0.f;
  if (idx < 9216) wsGORI[idx] = 0.f;
}

// ---------------------------------------------------------------------------
// Kernel: pack weights as f16 [N][K] (B-fragment friendly) incl. transposes.
// ---------------------------------------------------------------------------
__global__ __launch_bounds__(256) void k_prep_weights(
    const float* pw0, const float* pw1, const float* ew0, const float* ew1,
    half_t* wPW0, half_t* wPW1, half_t* wPW1T, half_t* wPW0T, half_t* wEW0,
    half_t* wEW1, half_t* wEW1T, half_t* wEW0T) {
  int idx = blockIdx.x * 256 + threadIdx.x;
  if (idx < 4096) {  // pw0 fwd B: [n=128][k=32], k<15 real
    int n = idx >> 5, k = idx & 31;
    wPW0[idx] = (k < 15) ? (half_t)pw0[k * 128 + n] : (half_t)0.f;
    return;
  }
  idx -= 4096;
  if (idx < 16384) {  // pw1 fwd B: [n=128][k=128]
    int n = idx >> 7, k = idx & 127;
    wPW1[idx] = (half_t)pw1[k * 128 + n];
    return;
  }
  idx -= 16384;
  if (idx < 16384) {  // pw1^T B: Weff[k][n]=pw1[n][k] -> [n][k]=pw1[n*128+k]
    int n = idx >> 7, k = idx & 127;
    wPW1T[idx] = (half_t)pw1[n * 128 + k];
    return;
  }
  idx -= 16384;
  if (idx < 2048) {  // pw0^T B: [n=16][k=128], n<15 real
    int n = idx >> 7, k = idx & 127;
    wPW0T[idx] = (n < 15) ? (half_t)pw0[n * 128 + k] : (half_t)0.f;
    return;
  }
  idx -= 2048;
  if (idx < 65536) {  // ew0 fwd B: [n=256][k=256], k<240 real
    int n = idx >> 8, k = idx & 255;
    wEW0[idx] = (k < 240) ? (half_t)ew0[k * 256 + n] : (half_t)0.f;
    return;
  }
  idx -= 65536;
  if (idx < 65536) {  // ew1 fwd B
    int n = idx >> 8, k = idx & 255;
    wEW1[idx] = (half_t)ew1[k * 256 + n];
    return;
  }
  idx -= 65536;
  if (idx < 65536) {  // ew1^T B
    int n = idx >> 8, k = idx & 255;
    wEW1T[idx] = (half_t)ew1[n * 256 + k];
    return;
  }
  idx -= 65536;
  if (idx < 65536) {  // ew0^T B: [n=256][k=256], n<240 real
    int n = idx >> 8, k = idx & 255;
    wEW0T[idx] = (n < 240) ? (half_t)ew0[n * 256 + k] : (half_t)0.f;
  }
}

// ---------------------------------------------------------------------------
// Kernel: pair forward. 2 blocks per batch row (64 pairs each).
// Produces DES[b,240] (padded 256) and prior_sum[b] via atomics.
// ---------------------------------------------------------------------------
__global__ __launch_bounds__(256) void k_pair_forward(
    const float* __restrict__ dr, const float* __restrict__ ori,
    const float* __restrict__ nori, const float* __restrict__ pb0,
    const float* __restrict__ pb1, const float* __restrict__ pw2,
    const float* __restrict__ pb2, const float* __restrict__ f1,
    const float* __restrict__ f2, const half_t* __restrict__ wPW0,
    const half_t* __restrict__ wPW1, float* __restrict__ wsDES,
    float* __restrict__ wsPS) {
  __shared__ __align__(16) half_t smF[64 * 32];
  __shared__ __align__(16) half_t smH0[64 * 128];
  __shared__ __align__(16) half_t smH1[64 * 128];
  __shared__ float smPfs[64 * 12];
  __shared__ float smPreF[64 * 20];
  __shared__ float smPrior;

  const int b = blockIdx.x >> 1;
  const int n0 = (blockIdx.x & 1) * 64;
  const int t = threadIdx.x;

  const float zc[6] = {0.5f, 0.125f, 7.8125e-3f, 3.0517578125e-5f,
                       4.656612873077393e-10f, 1.0842021724855044e-19f};

  if (t == 64) smPrior = 0.f;

  float R = 0.f;
  if (t < 64) {
    const int n = n0 + t;
    const size_t ib = (size_t)b * 128 + n;
    float dx = dr[ib * 3 + 0], dy = dr[ib * 3 + 1], dz = dr[ib * 3 + 2];
    R = sqrtf(dx * dx + dy * dy + dz * dz);
    float inv = 1.f / R;
    float ux = dx * inv, uy = dy * inv, uz = dz * inv;
    float o[9], no[9];
#pragma unroll
    for (int i = 0; i < 9; ++i) {
      o[i] = ori[(size_t)b * 9 + i];
      no[i] = nori[ib * 9 + i];
    }
    float f[15];
#pragma unroll
    for (int k = 0; k < 3; ++k) {
      f[k] = ux * o[k] + uy * o[3 + k] + uz * o[6 + k];
      f[3 + k] = ux * no[k] + uy * no[3 + k] + uz * no[6 + k];
    }
#pragma unroll
    for (int j = 0; j < 3; ++j)
#pragma unroll
      for (int k = 0; k < 3; ++k)
        f[6 + 3 * j + k] =
            o[j] * no[k] + o[3 + j] * no[3 + k] + o[6 + j] * no[6 + k];
#pragma unroll
    for (int i = 0; i < 15; ++i) smF[t * 32 + i] = (half_t)f[i];
#pragma unroll
    for (int i = 15; i < 32; ++i) smF[t * 32 + i] = (half_t)0.f;

    float fcR = __cosf(PI_F * R / RCUT) * 0.5f + 0.5f;
#pragma unroll
    for (int r = 0; r < 10; ++r) {
      float d = R - (3.f + (float)r * (2.f / 3.f));
      smPreF[t * 20 + r * 2 + 0] = __expf(-2.f * d * d) * fcR;  // eta=2
      smPreF[t * 20 + r * 2 + 1] = __expf(-1.f * d * d) * fcR;  // eta=1
    }
    float s0[6] = {0, 0, 0, 0, 0, 0}, s1a[6] = {0, 0, 0, 0, 0, 0};
#pragma unroll
    for (int i = 0; i < 15; ++i) {
      float vm = 1.f - f[i], vp = 1.f + f[i];
      float pm = vm * vm, pp = vp * vp;  // (1∓a)^2, squared each zeta step
#pragma unroll
      for (int z = 0; z < 6; ++z) {
        s0[z] += pm;
        s1a[z] += pp;
        pm *= pm;
        pp *= pp;
      }
    }
#pragma unroll
    for (int z = 0; z < 6; ++z) {
      smPfs[t * 12 + z * 2 + 0] = zc[z] * s0[z];
      smPfs[t * 12 + z * 2 + 1] = zc[z] * s1a[z];
    }
  }
  __syncthreads();

  prior_mlp_fwd((lds_chp)smF, (lds_hp)smH0, (lds_hp)smH1, (glb_chp)wPW0,
                (glb_chp)wPW1, pb0, pb1);

  if (t < 64) {  // output layer + prior
    float m_out = pb2[0];
    for (int h8 = 0; h8 < 128; h8 += 8) {
      v8h hv = *(const ASL v8h*)((lds_chp)smH1 + t * 128 + h8);
#pragma unroll
      for (int i = 0; i < 8; ++i) m_out += (float)hv[i] * pw2[h8 + i];
    }
    float e0 = fast_tanh(m_out);
    float base = R - (e0 * e0 + EPSF);
    float pr = 0.f;
#pragma unroll
    for (int q = 0; q < 3; ++q) {
      float A1 = f1[q] * f1[q] + EPSF, A2 = f2[q] * f2[q] + EPSF;
      pr += __powf(base * A1, -A2);
    }
    atomicAdd(&smPrior, pr);
  }
  __syncthreads();
  if (t == 0) atomicAdd(&wsPS[b], smPrior);
  if (t < 240) {  // DES rank-1 reduction: fcr = preF ⊗ pfs
    int z = t / 40, rem = t % 40, r = rem / 4, cc = rem % 4;
    int c = cc >> 1, e = cc & 1;
    float acc = 0.f;
    for (int n = 0; n < 64; ++n)
      acc += smPreF[n * 20 + r * 2 + e] * smPfs[n * 12 + z * 2 + c];
    atomicAdd(&wsDES[(size_t)b * 256 + t], acc);
  }
}

// ---------------------------------------------------------------------------
// Kernel: energy MLP forward (16 rows per block), saves h0e/h1e, writes E.
// ---------------------------------------------------------------------------
__global__ __launch_bounds__(256) void k_energy_fwd(
    const float* __restrict__ des, const float* __restrict__ eb0,
    const float* __restrict__ eb1, const float* __restrict__ ew2,
    const float* __restrict__ eb2, const float* __restrict__ wsPS,
    const half_t* __restrict__ wEW0, const half_t* __restrict__ wEW1,
    half_t* __restrict__ h0e, half_t* __restrict__ h1e,
    float* __restrict__ outE, float* __restrict__ wsGPS,
    float* __restrict__ wsGS1) {
  __shared__ __align__(16) half_t sH0[16 * 256];
  __shared__ __align__(16) half_t sH1[16 * 256];
  const int m0 = blockIdx.x * 16;
  const int t = threadIdx.x, lane = t & 31, wave = t >> 5;
  const int nt0 = wave * 2, hi = lane >> 4;
  {  // layer 0: K=240 (padded 256 in ws DES)
    v8f c0 = vzero8(), c1 = vzero8();
#pragma unroll
    for (int kc = 0; kc < 8; ++kc) {
      v16h a = loadA_gf32((glb_cfp)des, 256, m0, 32 * kc, lane);
      v16h b0 = loadB_pk((glb_chp)wEW0, 256, 32 * kc, 16 * nt0, lane);
      c0 = wmma16(a, b0, c0);
      v16h b1 = loadB_pk((glb_chp)wEW0, 256, 32 * kc, 16 * nt0 + 16, lane);
      c1 = wmma16(a, b1, c1);
    }
    int nn0 = 16 * nt0 + (lane & 15), nn1 = nn0 + 16;
    float bias0 = eb0[nn0], bias1 = eb0[nn1];
#pragma unroll
    for (int v = 0; v < 8; ++v) {
      int m = v + 8 * hi;
      half_t t0 = (half_t)fast_tanh(c0[v] + bias0);
      half_t t1 = (half_t)fast_tanh(c1[v] + bias1);
      ((lds_hp)sH0)[m * 256 + nn0] = t0;
      ((glb_hp)h0e)[(size_t)(m0 + m) * 256 + nn0] = t0;
      ((lds_hp)sH0)[m * 256 + nn1] = t1;
      ((glb_hp)h0e)[(size_t)(m0 + m) * 256 + nn1] = t1;
    }
  }
  __syncthreads();
  {  // layer 1
    v8f c0 = vzero8(), c1 = vzero8();
#pragma unroll
    for (int kc = 0; kc < 8; ++kc) {
      v16h a = loadA_lds((lds_chp)sH0, 256, 0, 32 * kc, lane);
      v16h b0 = loadB_pk((glb_chp)wEW1, 256, 32 * kc, 16 * nt0, lane);
      c0 = wmma16(a, b0, c0);
      v16h b1 = loadB_pk((glb_chp)wEW1, 256, 32 * kc, 16 * nt0 + 16, lane);
      c1 = wmma16(a, b1, c1);
    }
    int nn0 = 16 * nt0 + (lane & 15), nn1 = nn0 + 16;
    float bias0 = eb1[nn0], bias1 = eb1[nn1];
#pragma unroll
    for (int v = 0; v < 8; ++v) {
      int m = v + 8 * hi;
      half_t t0 = (half_t)fast_tanh(c0[v] + bias0);
      half_t t1 = (half_t)fast_tanh(c1[v] + bias1);
      ((lds_hp)sH1)[m * 256 + nn0] = t0;
      ((glb_hp)h1e)[(size_t)(m0 + m) * 256 + nn0] = t0;
      ((lds_hp)sH1)[m * 256 + nn1] = t1;
      ((glb_hp)h1e)[(size_t)(m0 + m) * 256 + nn1] = t1;
    }
  }
  __syncthreads();
  if (t < 16) {
    int row = m0 + t;
    float s1v = eb2[0];
    for (int h8 = 0; h8 < 256; h8 += 8) {
      v8h hv = *(const ASL v8h*)((lds_chp)sH1 + t * 256 + h8);
#pragma unroll
      for (int i = 0; i < 8; ++i) s1v += (float)hv[i] * ew2[h8 + i];
    }
    float sym = (s1v > 0.f) ? s1v : SLOPE * s1v;
    float tt = sym + wsPS[row];
    outE[row] = (tt > 0.f) ? tt : SLOPE * tt;
    float gt = (tt > 0.f) ? 1.f : SLOPE;
    wsGPS[row] = gt;
    wsGS1[row] = gt * ((s1v > 0.f) ? 1.f : SLOPE);
  }
}

// ---------------------------------------------------------------------------
// Kernel: energy backward part 1: gz0 = (gz1 @ ew1^T) * (1 - h0e^2),
// gz1 built on the fly (vector loads of h1e and ew2).
// ---------------------------------------------------------------------------
__global__ __launch_bounds__(256) void k_energy_bwd_gz0(
    const half_t* __restrict__ h1e, const half_t* __restrict__ h0e,
    const float* __restrict__ gs1, const float* __restrict__ ew2,
    const half_t* __restrict__ wEW1T, half_t* __restrict__ gz0e) {
  const int m0 = blockIdx.x * 16;
  const int t = threadIdx.x, lane = t & 31, wave = t >> 5;
  const int nt0 = wave * 2, hi = lane >> 4;
  const int row = m0 + (lane & 15);
  const float g = gs1[row];
  v8f c0 = vzero8(), c1 = vzero8();
#pragma unroll
  for (int kc = 0; kc < 8; ++kc) {
    const int k0 = 32 * kc + (hi ? 8 : 0);
    glb_chp ph = (glb_chp)h1e + (size_t)row * 256 + k0;
    v8h hlo = *(const ASG v8h*)ph;
    v8h hup = *(const ASG v8h*)(ph + 16);
    glb_cfp pw = (glb_cfp)ew2 + k0;
    v4f w0 = *(const ASG v4f*)pw;
    v4f w1 = *(const ASG v4f*)(pw + 4);
    v4f w2 = *(const ASG v4f*)(pw + 16);
    v4f w3 = *(const ASG v4f*)(pw + 20);
    v16h a;
#pragma unroll
    for (int i = 0; i < 4; ++i) {
      float hA = (float)hlo[i], hB = (float)hlo[4 + i];
      float hC = (float)hup[i], hD = (float)hup[4 + i];
      a[i] = (half_t)(g * w0[i] * (1.f - hA * hA));
      a[4 + i] = (half_t)(g * w1[i] * (1.f - hB * hB));
      a[8 + i] = (half_t)(g * w2[i] * (1.f - hC * hC));
      a[12 + i] = (half_t)(g * w3[i] * (1.f - hD * hD));
    }
    v16h b0 = loadB_pk((glb_chp)wEW1T, 256, 32 * kc, 16 * nt0, lane);
    c0 = wmma16(a, b0, c0);
    v16h b1 = loadB_pk((glb_chp)wEW1T, 256, 32 * kc, 16 * nt0 + 16, lane);
    c1 = wmma16(a, b1, c1);
  }
#pragma unroll
  for (int v = 0; v < 8; ++v) {
    int m = m0 + v + 8 * hi;
    int nn0 = 16 * nt0 + (lane & 15), nn1 = nn0 + 16;
    float h00 = (float)h0e[(size_t)m * 256 + nn0];
    float h01 = (float)h0e[(size_t)m * 256 + nn1];
    ((glb_hp)gz0e)[(size_t)m * 256 + nn0] = (half_t)(c0[v] * (1.f - h00 * h00));
    ((glb_hp)gz0e)[(size_t)m * 256 + nn1] = (half_t)(c1[v] * (1.f - h01 * h01));
  }
}

// ---------------------------------------------------------------------------
// Kernel: energy backward part 2: gDES = gz0 @ ew0^T (padded 256 cols).
// ---------------------------------------------------------------------------
__global__ __launch_bounds__(256) void k_energy_bwd_gdes(
    const half_t* __restrict__ gz0e, const half_t* __restrict__ wEW0T,
    float* __restrict__ gdes) {
  const int m0 = blockIdx.x * 16;
  const int t = threadIdx.x, lane = t & 31, wave = t >> 5;
  const int nt0 = wave * 2, hi = lane >> 4;
  v8f c0 = vzero8(), c1 = vzero8();
#pragma unroll
  for (int kc = 0; kc < 8; ++kc) {
    v16h a = loadA_g16((glb_chp)gz0e, 256, m0, 32 * kc, lane);
    v16h b0 = loadB_pk((glb_chp)wEW0T, 256, 32 * kc, 16 * nt0, lane);
    c0 = wmma16(a, b0, c0);
    v16h b1 = loadB_pk((glb_chp)wEW0T, 256, 32 * kc, 16 * nt0 + 16, lane);
    c1 = wmma16(a, b1, c1);
  }
#pragma unroll
  for (int v = 0; v < 8; ++v) {
    int m = m0 + v + 8 * hi;
    int nn0 = 16 * nt0 + (lane & 15), nn1 = nn0 + 16;
    ((glb_fp)gdes)[(size_t)m * 256 + nn0] = c0[v];
    ((glb_fp)gdes)[(size_t)m * 256 + nn1] = c1[v];
  }
}

// ---------------------------------------------------------------------------
// Kernel: pair backward. Recomputes prior forward, analytic chain rule,
// two WMMA backward matmuls, force + g_ori accumulation.
// ---------------------------------------------------------------------------
__global__ __launch_bounds__(256) void k_pair_backward(
    const float* __restrict__ dr, const float* __restrict__ ori,
    const float* __restrict__ nori, const float* __restrict__ pb0,
    const float* __restrict__ pb1, const float* __restrict__ pw2,
    const float* __restrict__ pb2, const float* __restrict__ f1,
    const float* __restrict__ f2, const half_t* __restrict__ wPW0,
    const half_t* __restrict__ wPW1, const half_t* __restrict__ wPW1T,
    const half_t* __restrict__ wPW0T, const float* __restrict__ gdes,
    const float* __restrict__ gps, float* __restrict__ outF,
    float* __restrict__ wsGORI) {
  __shared__ __align__(16) half_t smF[64 * 32];
  __shared__ __align__(16) half_t smH0[64 * 128];
  __shared__ __align__(16) half_t smH1[64 * 128];
  __shared__ __align__(16) half_t smBuf[64 * 128];  // gz1p, then gz0
  __shared__ float smGD[240];
  __shared__ float smGm[64];
  __shared__ float smGF[64 * 16];
  __shared__ float sFor[3];
  __shared__ float sGo[9];

  const int b = blockIdx.x >> 1;
  const int n0 = (blockIdx.x & 1) * 64;
  const int t = threadIdx.x, lane = t & 31, wave = t >> 5;

  const float zc[6] = {0.5f, 0.125f, 7.8125e-3f, 3.0517578125e-5f,
                       4.656612873077393e-10f, 1.0842021724855044e-19f};
  const float zeta[6] = {2.f, 4.f, 8.f, 16.f, 32.f, 64.f};

  if (t < 240) smGD[t] = gdes[(size_t)b * 256 + t];
  if (t >= 240 && t < 243) sFor[t - 240] = 0.f;
  if (t >= 243 && t < 252) sGo[t - 243] = 0.f;

  float R = 0.f, fcR = 0.f, ux = 0.f, uy = 0.f, uz = 0.f;
  float f[15], o[9], no[9], pre[20], pfs[12];
  if (t < 64) {  // recompute phase-1
    const int n = n0 + t;
    const size_t ib = (size_t)b * 128 + n;
    float dx = dr[ib * 3 + 0], dy = dr[ib * 3 + 1], dz = dr[ib * 3 + 2];
    R = sqrtf(dx * dx + dy * dy + dz * dz);
    float inv = 1.f / R;
    ux = dx * inv;
    uy = dy * inv;
    uz = dz * inv;
#pragma unroll
    for (int i = 0; i < 9; ++i) {
      o[i] = ori[(size_t)b * 9 + i];
      no[i] = nori[ib * 9 + i];
    }
#pragma unroll
    for (int k = 0; k < 3; ++k) {
      f[k] = ux * o[k] + uy * o[3 + k] + uz * o[6 + k];
      f[3 + k] = ux * no[k] + uy * no[3 + k] + uz * no[6 + k];
    }
#pragma unroll
    for (int j = 0; j < 3; ++j)
#pragma unroll
      for (int k = 0; k < 3; ++k)
        f[6 + 3 * j + k] =
            o[j] * no[k] + o[3 + j] * no[3 + k] + o[6 + j] * no[6 + k];
#pragma unroll
    for (int i = 0; i < 15; ++i) smF[t * 32 + i] = (half_t)f[i];
#pragma unroll
    for (int i = 15; i < 32; ++i) smF[t * 32 + i] = (half_t)0.f;

    fcR = __cosf(PI_F * R / RCUT) * 0.5f + 0.5f;
#pragma unroll
    for (int r = 0; r < 10; ++r) {
      float d = R - (3.f + (float)r * (2.f / 3.f));
      pre[r * 2 + 0] = __expf(-2.f * d * d);
      pre[r * 2 + 1] = __expf(-1.f * d * d);
    }
    float s0[6] = {0, 0, 0, 0, 0, 0}, s1a[6] = {0, 0, 0, 0, 0, 0};
#pragma unroll
    for (int i = 0; i < 15; ++i) {
      float vm = 1.f - f[i], vp = 1.f + f[i];
      float pm = vm * vm, pp = vp * vp;
#pragma unroll
      for (int z = 0; z < 6; ++z) {
        s0[z] += pm;
        s1a[z] += pp;
        pm *= pm;
        pp *= pp;
      }
    }
#pragma unroll
    for (int z = 0; z < 6; ++z) {
      pfs[z * 2 + 0] = zc[z] * s0[z];
      pfs[z * 2 + 1] = zc[z] * s1a[z];
    }
  }
  __syncthreads();

  prior_mlp_fwd((lds_chp)smF, (lds_hp)smH0, (lds_hp)smH1, (glb_chp)wPW0,
                (glb_chp)wPW1, pb0, pb1);

  float g_R = 0.f;
  float gpfs[12];
  if (t < 64) {
    float m_out = pb2[0];
    for (int h8 = 0; h8 < 128; h8 += 8) {
      v8h hv = *(const ASL v8h*)((lds_chp)smH1 + t * 128 + h8);
#pragma unroll
      for (int i = 0; i < 8; ++i) m_out += (float)hv[i] * pw2[h8 + i];
    }
    float e0 = fast_tanh(m_out);
    float base = R - (e0 * e0 + EPSF);
    float gE = gps[b];
    float dpb = 0.f;
#pragma unroll
    for (int q = 0; q < 3; ++q) {
      float A1 = f1[q] * f1[q] + EPSF, A2 = f2[q] * f2[q] + EPSF;
      dpb += -A2 * A1 * __powf(base * A1, -A2 - 1.f);
    }
    float g_base = gE * dpb;
    g_R = g_base;  // prior side: base = R - enc
    float g_m = (-g_base) * 2.f * e0 * (1.f - e0 * e0);
    smGm[t] = g_m;

    // symmetry side: g_pfs and g_R through pre*fcR
#pragma unroll
    for (int z = 0; z < 6; ++z)
#pragma unroll
      for (int c = 0; c < 2; ++c) {
        float s = 0.f;
        for (int r = 0; r < 10; ++r)
#pragma unroll
          for (int e = 0; e < 2; ++e)
            s += smGD[z * 40 + r * 4 + c * 2 + e] * pre[r * 2 + e];
        gpfs[z * 2 + c] = s * fcR;
      }
    float dfc = -0.5f * (PI_F / RCUT) * __sinf(PI_F * R / RCUT);
    for (int r = 0; r < 10; ++r) {
      float d = R - (3.f + (float)r * (2.f / 3.f));
#pragma unroll
      for (int e = 0; e < 2; ++e) {
        float gpf = 0.f;
#pragma unroll
        for (int z = 0; z < 6; ++z)
#pragma unroll
          for (int c = 0; c < 2; ++c)
            gpf += smGD[z * 40 + r * 4 + c * 2 + e] * pfs[z * 2 + c];
        float eta = (e == 0) ? 2.f : 1.f;
        g_R += gpf * pre[r * 2 + e] * (-2.f * eta * d * fcR + dfc);
      }
    }
  }
  __syncthreads();

  // gz1p[m][h] = g_m[m] * pw2[h] * (1 - h1^2); vectorized 8-wide runs
  for (int rid = t; rid < 64 * 16; rid += 256) {
    int m = rid >> 4, off = (rid & 15) * 8;
    v8h hv = *(const ASL v8h*)((lds_chp)smH1 + m * 128 + off);
    v4f w0 = *(const ASG v4f*)((glb_cfp)pw2 + off);
    v4f w1 = *(const ASG v4f*)((glb_cfp)pw2 + off + 4);
    float gm = smGm[m];
    v8h out;
#pragma unroll
    for (int i = 0; i < 4; ++i) {
      float hA = (float)hv[i], hB = (float)hv[4 + i];
      out[i] = (half_t)(gm * w0[i] * (1.f - hA * hA));
      out[4 + i] = (half_t)(gm * w1[i] * (1.f - hB * hB));
    }
    *(ASL v8h*)((lds_hp)smBuf + m * 128 + off) = out;
  }
  __syncthreads();

  // gh0 = gz1p @ pw1^T ; gz0 = gh0 * (1 - h0^2), held in regs until all
  // waves finish reading gz1p, then written back into smBuf.
  {
    const int mt = wave >> 1, ntB = (wave & 1) * 4;
    float cg[4][8];
    v8f acc[4] = {vzero8(), vzero8(), vzero8(), vzero8()};
#pragma unroll
    for (int kc = 0; kc < 4; ++kc) {
      v16h a = loadA_lds((lds_chp)smBuf, 128, 16 * mt, 32 * kc, lane);
#pragma unroll
      for (int j = 0; j < 4; ++j) {
        v16h bfr = loadB_pk((glb_chp)wPW1T, 128, 32 * kc, 16 * (ntB + j), lane);
        acc[j] = wmma16(a, bfr, acc[j]);
      }
    }
#pragma unroll
    for (int j = 0; j < 4; ++j)
#pragma unroll
      for (int v = 0; v < 8; ++v) {
        int m = 16 * mt + v + 8 * (lane >> 4);
        int nn = 16 * (ntB + j) + (lane & 15);
        float h0v = (float)smH0[m * 128 + nn];
        cg[j][v] = acc[j][v] * (1.f - h0v * h0v);
      }
    __syncthreads();
#pragma unroll
    for (int j = 0; j < 4; ++j)
#pragma unroll
      for (int v = 0; v < 8; ++v) {
        int m = 16 * mt + v + 8 * (lane >> 4);
        int nn = 16 * (ntB + j) + (lane & 15);
        smBuf[m * 128 + nn] = (half_t)cg[j][v];
      }
  }
  __syncthreads();

  // g_feats_prior = gz0 @ pw0^T  (N=16, only waves 0..3)
  if (wave < 4) {
    v8f cfr = vzero8();
#pragma unroll
    for (int kc = 0; kc < 4; ++kc) {
      v16h a = loadA_lds((lds_chp)smBuf, 128, 16 * wave, 32 * kc, lane);
      v16h bfr = loadB_pk((glb_chp)wPW0T, 128, 32 * kc, 0, lane);
      cfr = wmma16(a, bfr, cfr);
    }
#pragma unroll
    for (int v = 0; v < 8; ++v) {
      int m = 16 * wave + v + 8 * (lane >> 4);
      smGF[m * 16 + (lane & 15)] = cfr[v];
    }
  }
  __syncthreads();

  if (t < 64) {  // combine, map to dr and orientation
    float gf[15];
#pragma unroll
    for (int i = 0; i < 15; ++i) {
      float a = f[i];
      float vm = 1.f - a, vp = 1.f + a;
      float qm = vm, qp = vp;          // (1∓a)^(zeta-1): 1,3,7,15,31,63
      float wm = vm * vm, wp = vp * vp;
      float g = 0.f;
#pragma unroll
      for (int z = 0; z < 6; ++z) {
        g += gpfs[z * 2 + 0] * (-zc[z] * zeta[z] * qm) +
             gpfs[z * 2 + 1] * (zc[z] * zeta[z] * qp);
        qm *= wm;
        qp *= wp;
        wm *= wm;
        wp *= wp;
      }
      gf[i] = g + smGF[t * 16 + i];
    }
    float gu[3];
#pragma unroll
    for (int c = 0; c < 3; ++c) {
      float s = 0.f;
#pragma unroll
      for (int k = 0; k < 3; ++k)
        s += gf[k] * o[c * 3 + k] + gf[3 + k] * no[c * 3 + k];
      gu[c] = s;
    }
    float u3[3] = {ux, uy, uz};
    float gdot = gu[0] * ux + gu[1] * uy + gu[2] * uz;
    float invR = 1.f / R;
#pragma unroll
    for (int d = 0; d < 3; ++d) {
      float gd = (gu[d] - gdot * u3[d]) * invR + g_R * u3[d];
      atomicAdd(&sFor[d], -gd);
    }
#pragma unroll
    for (int c = 0; c < 3; ++c)
#pragma unroll
      for (int k = 0; k < 3; ++k) {
        float s = gf[k] * u3[c];
#pragma unroll
        for (int k2 = 0; k2 < 3; ++k2) s += gf[6 + 3 * k + k2] * no[c * 3 + k2];
        atomicAdd(&sGo[c * 3 + k], s);
      }
  }
  __syncthreads();
  if (t < 3) atomicAdd(&outF[(size_t)b * 3 + t], sFor[t]);
  if (t >= 3 && t < 12) atomicAdd(&wsGORI[(size_t)b * 9 + (t - 3)], sGo[t - 3]);
}

// ---------------------------------------------------------------------------
// Kernel: torque = sum_i cross(-g_ori[:,i], ori[:,i])
// ---------------------------------------------------------------------------
__global__ __launch_bounds__(256) void k_torque(const float* __restrict__ wsGORI,
                                                const float* __restrict__ ori,
                                                float* __restrict__ outT) {
  int b = blockIdx.x * 256 + threadIdx.x;
  if (b >= 1024) return;
  float go[9], o[9];
#pragma unroll
  for (int i = 0; i < 9; ++i) {
    go[i] = wsGORI[(size_t)b * 9 + i];
    o[i] = ori[(size_t)b * 9 + i];
  }
  float tx = 0.f, ty = 0.f, tz = 0.f;
#pragma unroll
  for (int i = 0; i < 3; ++i) {
    float ax = -go[i], ay = -go[3 + i], az = -go[6 + i];
    float bx = o[i], by = o[3 + i], bz = o[6 + i];
    tx += ay * bz - az * by;
    ty += az * bx - ax * bz;
    tz += ax * by - ay * bx;
  }
  outT[(size_t)b * 3 + 0] = tx;
  outT[(size_t)b * 3 + 1] = ty;
  outT[(size_t)b * 3 + 2] = tz;
}

// ---------------------------------------------------------------------------
// Launch
// ---------------------------------------------------------------------------
extern "C" void kernel_launch(void* const* d_in, const int* in_sizes, int n_in,
                              void* d_out, int out_size, void* d_ws,
                              size_t ws_size, hipStream_t stream) {
  const float* dr   = (const float*)d_in[0];
  const float* orn  = (const float*)d_in[1];
  const float* nori = (const float*)d_in[2];
  const float* pw0  = (const float*)d_in[3];
  const float* pb0  = (const float*)d_in[4];
  const float* pw1  = (const float*)d_in[5];
  const float* pb1  = (const float*)d_in[6];
  const float* pw2  = (const float*)d_in[7];
  const float* pb2  = (const float*)d_in[8];
  const float* ew0  = (const float*)d_in[9];
  const float* eb0  = (const float*)d_in[10];
  const float* ew1  = (const float*)d_in[11];
  const float* eb1  = (const float*)d_in[12];
  const float* ew2  = (const float*)d_in[13];
  const float* eb2  = (const float*)d_in[14];
  const float* f1   = (const float*)d_in[15];
  const float* f2   = (const float*)d_in[16];

  float* outF = (float*)d_out;          // [1024,3]
  float* outT = outF + 3072;            // [1024,3]
  float* outE = outF + 6144;            // [1024,1]

  char* ws = (char*)d_ws;
  half_t* wPW0  = (half_t*)(ws + OFF_PW0);
  half_t* wPW1  = (half_t*)(ws + OFF_PW1);
  half_t* wPW1T = (half_t*)(ws + OFF_PW1T);
  half_t* wPW0T = (half_t*)(ws + OFF_PW0T);
  half_t* wEW0  = (half_t*)(ws + OFF_EW0);
  half_t* wEW1  = (half_t*)(ws + OFF_EW1);
  half_t* wEW1T = (half_t*)(ws + OFF_EW1T);
  half_t* wEW0T = (half_t*)(ws + OFF_EW0T);
  half_t* h0e   = (half_t*)(ws + OFF_H0E);
  half_t* h1e   = (half_t*)(ws + OFF_H1E);
  half_t* gz0e  = (half_t*)(ws + OFF_GZ0E);
  float* wsDES  = (float*)(ws + OFF_DES);
  float* wsGDES = (float*)(ws + OFF_GDES);
  float* wsPS   = (float*)(ws + OFF_PS);
  float* wsGPS  = (float*)(ws + OFF_GPS);
  float* wsGS1  = (float*)(ws + OFF_GS1);
  float* wsGORI = (float*)(ws + OFF_GORI);

  k_zero<<<1024, 256, 0, stream>>>(outF, wsDES, wsPS, wsGORI);
  k_prep_weights<<<1176, 256, 0, stream>>>(pw0, pw1, ew0, ew1, wPW0, wPW1,
                                           wPW1T, wPW0T, wEW0, wEW1, wEW1T,
                                           wEW0T);
  k_pair_forward<<<2048, 256, 0, stream>>>(dr, orn, nori, pb0, pb1, pw2, pb2,
                                           f1, f2, wPW0, wPW1, wsDES, wsPS);
  k_energy_fwd<<<64, 256, 0, stream>>>(wsDES, eb0, eb1, ew2, eb2, wsPS, wEW0,
                                       wEW1, h0e, h1e, outE, wsGPS, wsGS1);
  k_energy_bwd_gz0<<<64, 256, 0, stream>>>(h1e, h0e, wsGS1, ew2, wEW1T, gz0e);
  k_energy_bwd_gdes<<<64, 256, 0, stream>>>(gz0e, wEW0T, wsGDES);
  k_pair_backward<<<2048, 256, 0, stream>>>(dr, orn, nori, pb0, pb1, pw2, pb2,
                                            f1, f2, wPW0, wPW1, wPW1T, wPW0T,
                                            wsGDES, wsGPS, outF, wsGORI);
  k_torque<<<4, 256, 0, stream>>>(wsGORI, orn, outT);
}
